// Model_43430709297247
// MI455X (gfx1250) — compile-verified, matrix operand order checked by reading, not run on hardware
//
#include <hip/hip_runtime.h>
#include <hip/hip_bf16.h>

// ---------------------------------------------------------------------------
// CDNA5 (gfx1250) fused hetero-SAGE-LSTM GNN.
// All GEMMs run on v_wmma_f32_16x16x32_bf16 (bf16 in, f32 accumulate).
// Gate weights are packed with (unit x gate)-interleaved columns so the LSTM
// nonlinearities are applied per 16x16 WMMA tile through a 1KB LDS buffer,
// keeping LDS/wave low (~24KB) for occupancy.
// ---------------------------------------------------------------------------

typedef __attribute__((ext_vector_type(16))) __bf16 v16bf;
typedef __attribute__((ext_vector_type(8)))  __bf16 v8bf;
typedef __attribute__((ext_vector_type(8)))  float  v8f;

#define NREL 3
#define DNBR 8

// ISA 7.12.2: 16-bit A-matrix 16x32 element->K map for lane L, element e:
//   K = (e&8)*2 + ((L>=16)?8:0) + (e&7)
__device__ __forceinline__ int kmap(int e, int lane) {
  return ((e & 8) << 1) | ((lane >> 4) << 3) | (e & 7);
}

__device__ __forceinline__ float sigmoidf_(float x) {
  return 1.0f / (1.0f + __expf(-x));
}

__device__ __forceinline__ v8f wmma_bf16(v16bf a, v16bf b, v8f c) {
  // (neg_a, A, neg_b, B, c_mod, C, reuse_a, reuse_b)
  return __builtin_amdgcn_wmma_f32_16x16x32_bf16(false, a, false, b,
                                                 (short)0, c, false, false);
}

// A fragment = two contiguous 16B runs of the LDS row -> two ds_load_b128.
__device__ __forceinline__ v16bf load_a_frag(const __bf16* rowp, int kt, int half) {
  const v8bf lo = *(const v8bf*)(rowp + kt * 32 + half * 8);
  const v8bf hi = *(const v8bf*)(rowp + kt * 32 + 16 + half * 8);
  return __builtin_shufflevector(lo, hi, 0, 1, 2, 3, 4, 5, 6, 7,
                                 8, 9, 10, 11, 12, 13, 14, 15);
}

// ---------------------------------------------------------------------------
// Pack row-major fp32 W[R][rows][cols] (used as B = W^T) into WMMA B-fragment
// order: frag (ct,kt), lane L holds output column n = ct*16 + L%16, element e
// holds K = kt*32 + kmap(e,L). Zero-padded outside rows/cols.
// gateI > 0: LSTM gate interleave -- tile ct's 16 columns are
//   (unit u = ct*4 + n/4, gate g = n%4) -> original row g*gateI + u,
// so one 16-col tile carries all 4 gates of 4 hidden units.
// ---------------------------------------------------------------------------
__global__ __launch_bounds__(32)
void pack_b(const float* __restrict__ W, __bf16* __restrict__ out,
            int rows, int cols, int nKt, int packElems, int gateI) {
  const int r    = blockIdx.y;
  const int tIdx = blockIdx.x;            // ct*nKt + kt
  const int kt   = tIdx % nKt;
  const int ct   = tIdx / nKt;
  const int lane = threadIdx.x;
  const int n    = lane & 15;
  int row;
  bool valid;
  if (gateI > 0) {
    int u = ct * 4 + (n >> 2);
    int g = n & 3;
    row   = g * gateI + u;
    valid = (u < gateI);
  } else {
    row   = ct * 16 + n;
    valid = (row < rows);
  }
  const float* Wr = W + (size_t)r * rows * cols;
  __bf16* o = out + (size_t)r * packElems + ((size_t)tIdx * 32 + lane) * 16;
#pragma unroll
  for (int e = 0; e < 16; ++e) {
    int k = kt * 32 + kmap(e, lane);
    float v = (valid && k < cols) ? Wr[(size_t)row * cols + k] : 0.0f;
    o[e] = (__bf16)v;
  }
}

// ---------------------------------------------------------------------------
// Embedding concat -> x0 bf16 [N][64] (cols 46..63 zero).
// ---------------------------------------------------------------------------
__global__ void embed_concat(const int* __restrict__ conn_idx,
                             const float* __restrict__ conn_feat,
                             const float* __restrict__ e0,
                             const float* __restrict__ e1,
                             const float* __restrict__ e2,
                             __bf16* __restrict__ x0, int N) {
  int n = blockIdx.x * blockDim.x + threadIdx.x;
  if (n >= N) return;
  __bf16* o = x0 + (size_t)n * 64;
  int i0 = conn_idx[n * 3 + 0], i1 = conn_idx[n * 3 + 1], i2 = conn_idx[n * 3 + 2];
  int c = 0;
  for (int k = 0; k < 6; ++k)  o[c++] = (__bf16)e0[i0 * 6 + k];
  for (int k = 0; k < 2; ++k)  o[c++] = (__bf16)e1[i1 * 2 + k];
  for (int k = 0; k < 4; ++k)  o[c++] = (__bf16)e2[i2 * 4 + k];
  for (int k = 0; k < 34; ++k) o[c++] = (__bf16)conn_feat[(size_t)n * 34 + k];
  for (; c < 64; ++c)          o[c] = (__bf16)0.0f;
}

// ---------------------------------------------------------------------------
// One full hetero-SAGE layer: per relation, LSTM over 8 gathered neighbor
// rows, then self+neighbor projection + leaky-ReLU, mean over relations.
// One wave per 16-node tile; all state in LDS (~24KB/wave).
// ---------------------------------------------------------------------------
template<int I, int IPAD, int O, int OPADS, bool OUTF32>
__global__ __launch_bounds__(32)
void sage_layer(const __bf16* __restrict__ xin,      // [N][IPAD] zero-padded
                const int*    __restrict__ nbr,      // [R][N][D]
                const __bf16* __restrict__ WihP,     // packed [R][NGT*NKT*512]
                const __bf16* __restrict__ WhhP,
                const float*  __restrict__ bih,      // [R][4I]
                const float*  __restrict__ bhh,
                const __bf16* __restrict__ WselfP,   // packed [R][NOT*NKT*512]
                const __bf16* __restrict__ WneighP,
                const float*  __restrict__ bc,       // [R][O]
                void*         __restrict__ xout,     // bf16/f32 [N][OPADS]
                int N) {
  constexpr int NKT = IPAD / 32;            // K tiles
  constexpr int NGT = (4 * I + 15) / 16;    // gate col tiles (== ceil(I/4))
  constexpr int NOT = O / 16;               // projection col tiles
  constexpr int GATE_ELEMS = NGT * NKT * 512;
  constexpr int PROJ_ELEMS = NOT * NKT * 512;

  __shared__ __bf16 s_x  [16 * IPAD];  // this tile's rows
  __shared__ __bf16 s_xt [16 * IPAD];  // gathered neighbor rows (per step)
  __shared__ __bf16 s_hb [16 * IPAD];  // hidden state, bf16 for WMMA
  __shared__ float  s_c  [16 * IPAD];  // cell state, f32
  __shared__ float  s_gt [16 * 16];    // one gate WMMA tile
  __shared__ float  s_b  [4 * I];      // bih + bhh for current relation
  __shared__ float  s_out[16 * O];     // sum over relations

  const int lane  = threadIdx.x;
  const int row   = lane & 15;
  const int half  = lane >> 4;
  const int node0 = blockIdx.x * 16;

  // Load own rows (bf16, zero-padded to IPAD): b128 copies.
  {
    const uint4* s4 = (const uint4*)(xin + (size_t)(node0 + row) * IPAD +
                                     half * (IPAD / 2));
    uint4* d4 = (uint4*)(&s_x[row * IPAD + half * (IPAD / 2)]);
#pragma unroll
    for (int c4 = 0; c4 < IPAD / 16; ++c4) d4[c4] = s4[c4];
  }
  for (int j = lane; j < 16 * O; j += 32) s_out[j] = 0.0f;
  __syncthreads();

  // A fragments of own rows, reused across relations / col tiles.
  v16bf aX[NKT];
#pragma unroll
  for (int kt = 0; kt < NKT; ++kt)
    aX[kt] = load_a_frag(&s_x[row * IPAD], kt, half);

  for (int r = 0; r < NREL; ++r) {
    for (int j = lane; j < 16 * IPAD; j += 32) {
      s_c[j]  = 0.0f;
      s_hb[j] = (__bf16)0.0f;
    }
    for (int j = lane; j < 4 * I; j += 32)
      s_b[j] = bih[(size_t)r * 4 * I + j] + bhh[(size_t)r * 4 * I + j];
    __syncthreads();

    const __bf16* wih = WihP + (size_t)r * GATE_ELEMS;
    const __bf16* whh = WhhP + (size_t)r * GATE_ELEMS;

    for (int t = 0; t < DNBR; ++t) {
      // Gather neighbor rows for this timestep (b128 copies).
      int idx = nbr[((size_t)r * N + node0 + row) * DNBR + t];
      {
        const uint4* s4 = (const uint4*)(xin + (size_t)idx * IPAD +
                                         half * (IPAD / 2));
        uint4* d4 = (uint4*)(&s_xt[row * IPAD + half * (IPAD / 2)]);
#pragma unroll
        for (int c4 = 0; c4 < IPAD / 16; ++c4) d4[c4] = s4[c4];
      }
      // Prefetch next step's gather rows so they land in cache while the
      // WMMAs below run (gfx1250: global_prefetch_b8).
      if (t + 1 < DNBR) {
        int idxn = nbr[((size_t)r * N + node0 + row) * DNBR + t + 1];
        __builtin_prefetch(xin + (size_t)idxn * IPAD + half * (IPAD / 2), 0, 1);
      }
      __syncthreads();

      // A fragments for xt and h (hoisted out of the col-tile loop).
      v16bf aXT[NKT], aH[NKT];
#pragma unroll
      for (int kt = 0; kt < NKT; ++kt) {
        aXT[kt] = load_a_frag(&s_xt[row * IPAD], kt, half);
        aH [kt] = load_a_frag(&s_hb[row * IPAD], kt, half);
      }

      // Per gate tile: g = xt @ Wih^T + h @ Whh^T, then fuse the LSTM cell
      // update for the 4 hidden units held by this tile.
      for (int ct = 0; ct < NGT; ++ct) {
        v8f acc = {};
#pragma unroll
        for (int kt = 0; kt < NKT; ++kt) {
          v16bf b0 = *(const v16bf*)(wih + ((size_t)(ct * NKT + kt) * 32 + lane) * 16);
          acc = wmma_bf16(aXT[kt], b0, acc);
          v16bf b1 = *(const v16bf*)(whh + ((size_t)(ct * NKT + kt) * 32 + lane) * 16);
          acc = wmma_bf16(aH[kt], b1, acc);
        }
        // C/D layout: VGPR v -> M = v + 8*(lane>=16), N = lane%16.
#pragma unroll
        for (int v = 0; v < 8; ++v)
          s_gt[(v + half * 8) * 16 + row] = acc[v];
        __syncthreads();

        // 64 cell updates (16 rows x 4 units), 2 per lane.
#pragma unroll
        for (int e2 = 0; e2 < 2; ++e2) {
          int elem = lane + 32 * e2;    // 0..63
          int m    = elem & 15;
          int ul   = elem >> 4;         // unit within tile
          int u    = ct * 4 + ul;
          if (u < I) {
            float gi = s_gt[m * 16 + ul * 4 + 0] + s_b[0 * I + u];
            float gf = s_gt[m * 16 + ul * 4 + 1] + s_b[1 * I + u];
            float gg = s_gt[m * 16 + ul * 4 + 2] + s_b[2 * I + u];
            float go = s_gt[m * 16 + ul * 4 + 3] + s_b[3 * I + u];
            float c  = sigmoidf_(gf) * s_c[m * IPAD + u] +
                       sigmoidf_(gi) * tanhf(gg);
            float h  = sigmoidf_(go) * tanhf(c);
            s_c [m * IPAD + u] = c;
            s_hb[m * IPAD + u] = (__bf16)h;   // padded cols stay zero
          }
        }
        __syncthreads();
      }
    }

    // o = x @ Wself^T + h_n @ Wneigh^T + bc ; leaky ; accumulate.
    v16bf aHn[NKT];
#pragma unroll
    for (int kt = 0; kt < NKT; ++kt)
      aHn[kt] = load_a_frag(&s_hb[row * IPAD], kt, half);

    const __bf16* wself  = WselfP  + (size_t)r * PROJ_ELEMS;
    const __bf16* wneigh = WneighP + (size_t)r * PROJ_ELEMS;
    for (int ct = 0; ct < NOT; ++ct) {
      v8f acc = {};
#pragma unroll
      for (int kt = 0; kt < NKT; ++kt) {
        v16bf b0 = *(const v16bf*)(wself  + ((size_t)(ct * NKT + kt) * 32 + lane) * 16);
        acc = wmma_bf16(aX[kt], b0, acc);
        v16bf b1 = *(const v16bf*)(wneigh + ((size_t)(ct * NKT + kt) * 32 + lane) * 16);
        acc = wmma_bf16(aHn[kt], b1, acc);
      }
#pragma unroll
      for (int v = 0; v < 8; ++v) {
        int m   = v + half * 8;
        int col = ct * 16 + row;
        float val = acc[v] + bc[(size_t)r * O + col];
        val = val > 0.0f ? val : 0.01f * val;
        s_out[m * O + col] += val;
      }
    }
    __syncthreads();
  }

  // Mean over relations; store padded for next layer (or f32 for MLP head).
  const float inv3 = 1.0f / 3.0f;
  for (int j = lane; j < 16 * OPADS; j += 32) {
    int m = j / OPADS, cc = j - m * OPADS;
    float v = (cc < O) ? s_out[m * O + cc] * inv3 : 0.0f;
    if constexpr (OUTF32)
      ((float*)xout)[(size_t)(node0 + m) * OPADS + cc] = v;
    else
      ((__bf16*)xout)[(size_t)(node0 + m) * OPADS + cc] = (__bf16)v;
  }
}

// ---------------------------------------------------------------------------
// MLP head: 96 -> 72 (leaky) -> 10. ~0.5 GFLOP total; plain VALU.
// ---------------------------------------------------------------------------
__global__ void mlp_head(const float* __restrict__ x, const float* __restrict__ W1,
                         const float* __restrict__ b1, const float* __restrict__ W2,
                         const float* __restrict__ b2, float* __restrict__ out, int N) {
  int n = blockIdx.x * blockDim.x + threadIdx.x;
  if (n >= N) return;
  const float* xr = x + (size_t)n * 96;
  float t[72];
  for (int j = 0; j < 72; ++j) {
    float a = b1[j];
    for (int k = 0; k < 96; ++k) a += W1[j * 96 + k] * xr[k];
    t[j] = a > 0.0f ? a : 0.01f * a;
  }
  for (int c = 0; c < 10; ++c) {
    float a = b2[c];
    for (int k = 0; k < 72; ++k) a += W2[c * 72 + k] * t[k];
    out[(size_t)n * 10 + c] = a;
  }
}

// ---------------------------------------------------------------------------
extern "C" void kernel_launch(void* const* d_in, const int* in_sizes, int n_in,
                              void* d_out, int out_size, void* d_ws, size_t ws_size,
                              hipStream_t stream) {
  (void)n_in; (void)out_size; (void)ws_size;
  const int N = in_sizes[0] / 3;  // 32768

  const int*   conn_idx  = (const int*)d_in[0];
  const float* conn_feat = (const float*)d_in[1];
  const int*   nbr       = (const int*)d_in[2];
  const float* e0 = (const float*)d_in[3];
  const float* e1 = (const float*)d_in[4];
  const float* e2 = (const float*)d_in[5];
  const float* Wih1 = (const float*)d_in[6],  *Whh1 = (const float*)d_in[7];
  const float* bih1 = (const float*)d_in[8],  *bhh1 = (const float*)d_in[9];
  const float* Wse1 = (const float*)d_in[10], *Wne1 = (const float*)d_in[11];
  const float* bc1  = (const float*)d_in[12];
  const float* Wih2 = (const float*)d_in[13], *Whh2 = (const float*)d_in[14];
  const float* bih2 = (const float*)d_in[15], *bhh2 = (const float*)d_in[16];
  const float* Wse2 = (const float*)d_in[17], *Wne2 = (const float*)d_in[18];
  const float* bc2  = (const float*)d_in[19];
  const float* Wih3 = (const float*)d_in[20], *Whh3 = (const float*)d_in[21];
  const float* bih3 = (const float*)d_in[22], *bhh3 = (const float*)d_in[23];
  const float* Wse3 = (const float*)d_in[24], *Wne3 = (const float*)d_in[25];
  const float* bc3  = (const float*)d_in[26];
  const float* W1 = (const float*)d_in[27], *b1 = (const float*)d_in[28];
  const float* W2 = (const float*)d_in[29], *b2 = (const float*)d_in[30];

  // ---- workspace carving -------------------------------------------------
  char* ws = (char*)d_ws;
  size_t off = 0;
  auto alloc = [&](size_t bytes) -> void* {
    void* p = ws + off;
    off = (off + bytes + 255) & ~(size_t)255;
    return p;
  };
  __bf16* x0 = (__bf16*)alloc((size_t)N * 64 * 2);   // embed out, IPAD=64
  __bf16* x1 = (__bf16*)alloc((size_t)N * 64 * 2);   // layer1 out, IPAD=64
  __bf16* x2 = (__bf16*)alloc((size_t)N * 96 * 2);   // layer2 out, IPAD=96
  float*  x3 = (float*) alloc((size_t)N * 96 * 4);   // layer3 out, f32

  const int G1 = 12 * 2 * 512, P1 = 4 * 2 * 512;     // L1: I=46, 12 gate tiles
  const int G2 = 16 * 2 * 512, P2 = 5 * 2 * 512;     // L2: I=64
  const int G3 = 20 * 3 * 512, P3 = 6 * 3 * 512;     // L3: I=80, IPAD=96
  __bf16* wihP1 = (__bf16*)alloc((size_t)3 * G1 * 2);
  __bf16* whhP1 = (__bf16*)alloc((size_t)3 * G1 * 2);
  __bf16* wseP1 = (__bf16*)alloc((size_t)3 * P1 * 2);
  __bf16* wneP1 = (__bf16*)alloc((size_t)3 * P1 * 2);
  __bf16* wihP2 = (__bf16*)alloc((size_t)3 * G2 * 2);
  __bf16* whhP2 = (__bf16*)alloc((size_t)3 * G2 * 2);
  __bf16* wseP2 = (__bf16*)alloc((size_t)3 * P2 * 2);
  __bf16* wneP2 = (__bf16*)alloc((size_t)3 * P2 * 2);
  __bf16* wihP3 = (__bf16*)alloc((size_t)3 * G3 * 2);
  __bf16* whhP3 = (__bf16*)alloc((size_t)3 * G3 * 2);
  __bf16* wseP3 = (__bf16*)alloc((size_t)3 * P3 * 2);
  __bf16* wneP3 = (__bf16*)alloc((size_t)3 * P3 * 2);

  // ---- weight packing (fp32 -> bf16, WMMA B-fragment order) -------------
  // Gate matrices use unit/gate interleave (gateI = hidden size I).
  pack_b<<<dim3(24, 3), 32, 0, stream>>>(Wih1, wihP1, 184, 46, 2, G1, 46);
  pack_b<<<dim3(24, 3), 32, 0, stream>>>(Whh1, whhP1, 184, 46, 2, G1, 46);
  pack_b<<<dim3( 8, 3), 32, 0, stream>>>(Wse1, wseP1,  64, 46, 2, P1, 0);
  pack_b<<<dim3( 8, 3), 32, 0, stream>>>(Wne1, wneP1,  64, 46, 2, P1, 0);
  pack_b<<<dim3(32, 3), 32, 0, stream>>>(Wih2, wihP2, 256, 64, 2, G2, 64);
  pack_b<<<dim3(32, 3), 32, 0, stream>>>(Whh2, whhP2, 256, 64, 2, G2, 64);
  pack_b<<<dim3(10, 3), 32, 0, stream>>>(Wse2, wseP2,  80, 64, 2, P2, 0);
  pack_b<<<dim3(10, 3), 32, 0, stream>>>(Wne2, wneP2,  80, 64, 2, P2, 0);
  pack_b<<<dim3(60, 3), 32, 0, stream>>>(Wih3, wihP3, 320, 80, 3, G3, 80);
  pack_b<<<dim3(60, 3), 32, 0, stream>>>(Whh3, whhP3, 320, 80, 3, G3, 80);
  pack_b<<<dim3(18, 3), 32, 0, stream>>>(Wse3, wseP3,  96, 80, 3, P3, 0);
  pack_b<<<dim3(18, 3), 32, 0, stream>>>(Wne3, wneP3,  96, 80, 3, P3, 0);

  // ---- forward pass ------------------------------------------------------
  embed_concat<<<(N + 255) / 256, 256, 0, stream>>>(conn_idx, conn_feat,
                                                    e0, e1, e2, x0, N);

  sage_layer<46, 64, 64, 64, false><<<N / 16, 32, 0, stream>>>(
      x0, nbr, wihP1, whhP1, bih1, bhh1, wseP1, wneP1, bc1, (void*)x1, N);
  sage_layer<64, 64, 80, 96, false><<<N / 16, 32, 0, stream>>>(
      x1, nbr, wihP2, whhP2, bih2, bhh2, wseP2, wneP2, bc2, (void*)x2, N);
  sage_layer<80, 96, 96, 96, true><<<N / 16, 32, 0, stream>>>(
      x2, nbr, wihP3, whhP3, bih3, bhh3, wseP3, wneP3, bc3, (void*)x3, N);

  mlp_head<<<(N + 255) / 256, 256, 0, stream>>>(x3, W1, b1, W2, b2,
                                                (float*)d_out, N);
}
